// GAT_68307159875933
// MI455X (gfx1250) — matrix operand order repaired and checked by reading, not compiled, not run on hardware
//
#include <hip/hip_runtime.h>
#include <hip/hip_bf16.h>
#include <stdint.h>

#define N_NODES 50000
#define N_EDGES 800000
#define FEATS   64
#define HEADS   4
#define HD      256          // HEADS*FEATS
#define NEG_SLOPE 0.2f

typedef __attribute__((ext_vector_type(16))) __bf16 v16bf;
typedef __attribute__((ext_vector_type(8)))  __bf16 v8bf;
typedef __attribute__((ext_vector_type(8)))  float  v8f;

static __device__ __forceinline__ unsigned short f2bf_rne(float f) {
    unsigned u = __float_as_uint(f);
    unsigned r = u + 0x7FFFu + ((u >> 16) & 1u);   // round-to-nearest-even
    return (unsigned short)(r >> 16);
}

// float atomic max via monotone int/uint reinterpretation
static __device__ __forceinline__ void atomicMaxF32(float* addr, float val) {
    if (val >= 0.0f) atomicMax((int*)addr, __float_as_int(val));
    else             atomicMin((unsigned int*)addr, __float_as_uint(val));
}

// ---------------- init: zero out + denom, emax = -inf ----------------
__global__ void gat_init(float* __restrict__ out, float* __restrict__ emax,
                         float* __restrict__ denom) {
    int t = blockIdx.x * blockDim.x + threadIdx.x;
    if (t < N_NODES * FEATS) out[t] = 0.0f;
    if (t < N_NODES * HEADS) {
        emax[t]  = __int_as_float(0xFF800000);   // -inf
        denom[t] = 0.0f;
    }
}

// ---------------- x (f32) -> bf16 ----------------
__global__ void gat_cvt_x(const float* __restrict__ x, unsigned short* __restrict__ xh) {
    int t = blockIdx.x * blockDim.x + threadIdx.x;
    if (t < N_NODES * FEATS) xh[t] = f2bf_rne(x[t]);
}

// ---------------- pack W into bf16 B fragments, contiguous k order (k = 16*half + s) as A uses ----------------
// Wpack index = ((n_tile*2 + kblock)*32 + lane)*8 + j ; dword = {bf16(W[k0][n]), bf16(W[k0+1][n])}
// with k0 = kblock*32 + (lane>>4)*16 + 2*j, n = n_tile*16 + (lane&15)
__global__ void gat_pack_w(const float* __restrict__ W, uint32_t* __restrict__ Wpack) {
    int t = blockIdx.x * blockDim.x + threadIdx.x;   // 8192 threads
    int j    = t & 7;
    int lane = (t >> 3) & 31;
    int kb   = (t >> 8) & 1;
    int nt   = t >> 9;
    int n  = nt * 16 + (lane & 15);
    int k0 = kb * 32 + (lane >> 4) * 16 + 2 * j;
    unsigned lo = f2bf_rne(W[k0 * HD + n]);
    unsigned hi = f2bf_rne(W[(k0 + 1) * HD + n]);
    Wpack[t] = lo | (hi << 16);
}

// ---------------- WMMA GEMM: feat = x @ W  (bf16 x bf16 -> f32) ----------------
// one wave per 16x16 tile; K=64 -> two v_wmma_f32_16x16x32_bf16
__global__ void __launch_bounds__(256) gat_gemm(const unsigned short* __restrict__ xh,
                                                const uint32_t* __restrict__ Wpack,
                                                float* __restrict__ feat) {
    int gid  = blockIdx.x * 8 + (threadIdx.x >> 5);   // global wave id == tile id
    int lane = threadIdx.x & 31;
    int mt = gid >> 4;          // 3125 m-tiles
    int nt = gid & 15;          // 16 n-tiles
    int mbase = mt * 16;
    int hf  = lane >> 4;        // lane half
    int m16 = lane & 15;

    const __bf16* arow = (const __bf16*)(xh + (size_t)(mbase + m16) * FEATS);

    v16bf a0, a1;
    {
        v8bf lo = *(const v8bf*)(arow + hf * 16);
        v8bf hi = *(const v8bf*)(arow + 8 + hf * 16);
#pragma unroll
        for (int i = 0; i < 8; ++i) { a0[i] = lo[i]; a0[8 + i] = hi[i]; }
    }
    {
        v8bf lo = *(const v8bf*)(arow + 32 + hf * 16);
        v8bf hi = *(const v8bf*)(arow + 40 + hf * 16);
#pragma unroll
        for (int i = 0; i < 8; ++i) { a1[i] = lo[i]; a1[8 + i] = hi[i]; }
    }

    v16bf b0 = *(const v16bf*)(Wpack + ((size_t)(nt * 2 + 0) * 32 + lane) * 8);
    v16bf b1 = *(const v16bf*)(Wpack + ((size_t)(nt * 2 + 1) * 32 + lane) * 8);

    v8f c = {};
    c = __builtin_amdgcn_wmma_f32_16x16x32_bf16(false, a0, false, b0, (short)0, c, false, false);
    c = __builtin_amdgcn_wmma_f32_16x16x32_bf16(false, a1, false, b1, (short)0, c, false, false);

    // C/D layout: VGPR r, lanes 0-15 -> M=r ; lanes 16-31 -> M=8+r ; N = lane&15
    float* orow = feat + (size_t)(mbase + hf * 8) * HD + nt * 16 + m16;
#pragma unroll
    for (int r = 0; r < 8; ++r) orow[(size_t)r * HD] = c[r];
}

// ---------------- per (node, head) attention dots ----------------
__global__ void gat_scores(const float* __restrict__ feat, const float* __restrict__ attn_l,
                           const float* __restrict__ attn_r, float* __restrict__ el,
                           float* __restrict__ er) {
    int t = blockIdx.x * blockDim.x + threadIdx.x;
    if (t >= N_NODES * HEADS) return;
    int n = t >> 2, h = t & 3;
    const float* fr = feat + (size_t)n * HD + h * FEATS;
    const float* wl = attn_l + h * FEATS;
    const float* wr = attn_r + h * FEATS;
    float sl = 0.0f, sr = 0.0f;
#pragma unroll 8
    for (int d = 0; d < FEATS; ++d) {
        float f = fr[d];
        sl += f * wl[d];
        sr += f * wr[d];
    }
    el[t] = sl;
    er[t] = sr;
}

static __device__ __forceinline__ void edge_logits(const float4 a, const float4 b, float v[4]) {
    v[0] = a.x + b.x; v[1] = a.y + b.y; v[2] = a.z + b.z; v[3] = a.w + b.w;
#pragma unroll
    for (int h = 0; h < 4; ++h) v[h] = v[h] > 0.0f ? v[h] : NEG_SLOPE * v[h];
}

// ---------------- edge pass 1: segment max ----------------
__global__ void gat_edge_max(const int* __restrict__ src, const int* __restrict__ dst,
                             const float4* __restrict__ el4, const float4* __restrict__ er4,
                             float* __restrict__ emax) {
    int e = blockIdx.x * blockDim.x + threadIdx.x;
    if (e >= N_EDGES) return;
    int s = src[e], d = dst[e];
    float v[4];
    edge_logits(el4[s], er4[d], v);
#pragma unroll
    for (int h = 0; h < 4; ++h) atomicMaxF32(&emax[d * 4 + h], v[h]);
}

// ---------------- edge pass 2: segment sum of exp ----------------
__global__ void gat_edge_sum(const int* __restrict__ src, const int* __restrict__ dst,
                             const float4* __restrict__ el4, const float4* __restrict__ er4,
                             const float4* __restrict__ emax4, float* __restrict__ denom) {
    int e = blockIdx.x * blockDim.x + threadIdx.x;
    if (e >= N_EDGES) return;
    int s = src[e], d = dst[e];
    float v[4];
    edge_logits(el4[s], er4[d], v);
    float4 m = emax4[d];
    float mm[4] = {m.x, m.y, m.z, m.w};
#pragma unroll
    for (int h = 0; h < 4; ++h) atomicAdd(&denom[d * 4 + h], __expf(v[h] - mm[h]));
}

// ---------------- edge pass 3: alpha-weighted gather/scatter + head mean ----------------
// one wave per edge; lane l covers output dims {l, l+32}; head sum fused in registers
__global__ void __launch_bounds__(256) gat_edge_agg(const int* __restrict__ src,
                                                    const int* __restrict__ dst,
                                                    const float4* __restrict__ el4,
                                                    const float4* __restrict__ er4,
                                                    const float4* __restrict__ emax4,
                                                    const float4* __restrict__ denom4,
                                                    const float* __restrict__ feat,
                                                    float* __restrict__ out) {
    int e    = blockIdx.x * 8 + (threadIdx.x >> 5);
    int lane = threadIdx.x & 31;
    int s = src[e], d = dst[e];

    float v[4];
    edge_logits(el4[s], er4[d], v);
    float4 m  = emax4[d];
    float4 dn = denom4[d];
    float alpha[4];
    alpha[0] = __expf(v[0] - m.x) / dn.x;
    alpha[1] = __expf(v[1] - m.y) / dn.y;
    alpha[2] = __expf(v[2] - m.z) / dn.z;
    alpha[3] = __expf(v[3] - m.w) / dn.w;

    const float* fr = feat + (size_t)s * HD;
    __builtin_prefetch(fr + lane, 0, 1);   // global_prefetch_b8

    float acc0 = 0.0f, acc1 = 0.0f;
#pragma unroll
    for (int h = 0; h < 4; ++h) {
        acc0 += alpha[h] * fr[h * FEATS + lane];
        acc1 += alpha[h] * fr[h * FEATS + lane + 32];
    }
    atomicAdd(&out[(size_t)d * FEATS + lane],      0.25f * acc0);
    atomicAdd(&out[(size_t)d * FEATS + lane + 32], 0.25f * acc1);
}

extern "C" void kernel_launch(void* const* d_in, const int* in_sizes, int n_in,
                              void* d_out, int out_size, void* d_ws, size_t ws_size,
                              hipStream_t stream) {
    const float* x      = (const float*)d_in[0];
    const float* W      = (const float*)d_in[1];
    const float* attn_l = (const float*)d_in[2];
    const float* attn_r = (const float*)d_in[3];
    const int*   src    = (const int*)d_in[4];
    const int*   dst    = (const int*)d_in[5];
    float*       out    = (float*)d_out;

    char* ws = (char*)d_ws;
    size_t o = 0;
    float* feat  = (float*)(ws + o);           o += (size_t)N_NODES * HD * 4;      // 51.2 MB
    float* el    = (float*)(ws + o);           o += (size_t)N_NODES * HEADS * 4;   // 0.8 MB
    float* er    = (float*)(ws + o);           o += (size_t)N_NODES * HEADS * 4;
    float* emax  = (float*)(ws + o);           o += (size_t)N_NODES * HEADS * 4;
    float* denom = (float*)(ws + o);           o += (size_t)N_NODES * HEADS * 4;
    unsigned short* xh = (unsigned short*)(ws + o); o += (size_t)N_NODES * FEATS * 2; // 6.4 MB
    uint32_t* Wpack = (uint32_t*)(ws + o);     o += 16 * 2 * 32 * 8 * 4;           // 32 KB

    const int B = 256;
    gat_init   <<<(N_NODES * FEATS + B - 1) / B, B, 0, stream>>>(out, emax, denom);
    gat_cvt_x  <<<(N_NODES * FEATS + B - 1) / B, B, 0, stream>>>(x, xh);
    gat_pack_w <<<8192 / B, B, 0, stream>>>(W, Wpack);
    gat_gemm   <<<(N_NODES / 16) * 16 / 8, B, 0, stream>>>(xh, Wpack, feat);   // 6250 blocks
    gat_scores <<<(N_NODES * HEADS + B - 1) / B, B, 0, stream>>>(feat, attn_l, attn_r, el, er);
    gat_edge_max<<<(N_EDGES + B - 1) / B, B, 0, stream>>>(src, dst, (const float4*)el,
                                                          (const float4*)er, emax);
    gat_edge_sum<<<(N_EDGES + B - 1) / B, B, 0, stream>>>(src, dst, (const float4*)el,
                                                          (const float4*)er, (const float4*)emax,
                                                          denom);
    gat_edge_agg<<<N_EDGES / 8, B, 0, stream>>>(src, dst, (const float4*)el, (const float4*)er,
                                                (const float4*)emax, (const float4*)denom,
                                                feat, out);
}
